// MultiheadAttention_65859028517392
// MI455X (gfx1250) — matrix operand order repaired and checked by reading, not compiled, not run on hardware
//
#include <hip/hip_runtime.h>

typedef __attribute__((ext_vector_type(16))) _Float16 v16h;
typedef __attribute__((ext_vector_type(4)))  _Float16 v4h;
typedef __attribute__((ext_vector_type(8)))  float    v8f;
typedef __attribute__((ext_vector_type(4)))  int      i32x4;

#define D_MODEL 1024
#define N_HEAD  16
#define D_K     64
#define SEQ     2048
#define BATCH   2

// ---------------------------------------------------------------------------
// Async global->LDS copy of 16 bytes (8 f16) per lane. CDNA5 path uses the
// async copy engine (ASYNCcnt); fallback is a plain VGPR round-trip.
// ---------------------------------------------------------------------------
#if __has_builtin(__builtin_amdgcn_global_load_async_to_lds_b128)
#define ASYNC_LDS 1
#else
#define ASYNC_LDS 0
#endif

__device__ __forceinline__ void cp16_g2l(const _Float16* g, _Float16* l) {
#if ASYNC_LDS
  __builtin_amdgcn_global_load_async_to_lds_b128(
      (__attribute__((address_space(1))) i32x4*)g,
      (__attribute__((address_space(3))) i32x4*)l, 0, 0);
#else
  *(ulonglong2*)l = *(const ulonglong2*)g;
#endif
}

template <int N>
__device__ __forceinline__ void wait_async() {
#if __has_builtin(__builtin_amdgcn_s_wait_asynccnt)
  __builtin_amdgcn_s_wait_asynccnt(N);
#else
  asm volatile("s_wait_asynccnt %0" ::"n"(N) : "memory");
#endif
}

// ---------------------------------------------------------------------------
// fp32 -> f16 conversion (inputs and weights), vectorized x4
// ---------------------------------------------------------------------------
__global__ __launch_bounds__(256) void cvt_f32_f16(
    const float* __restrict__ in, _Float16* __restrict__ out, int n4)
{
  int i = blockIdx.x * 256 + threadIdx.x;
  if (i < n4) {
    float4 f = ((const float4*)in)[i];
    v4h h;
    h[0] = (_Float16)f.x; h[1] = (_Float16)f.y;
    h[2] = (_Float16)f.z; h[3] = (_Float16)f.w;
    ((v4h*)out)[i] = h;
  }
}

// ============================================================================
// GEMM: Y = X @ W^T   (X: [M,K] f16 row-major, W: [N,K] f16 row-major)
// mode 0: Y fp32 row-major [M,N]          (output projection -> d_out)
// mode 1: Y f16 scatter [B,H,S,d_k]       (Q,K projections)
// mode 2: Y f16 scatter [B,H,d_k,S]       (V projection, pre-transposed)
// Block 256 thr = 8 waves (4x2), tile 128x64, BK=32, double-buffered async LDS.
// ============================================================================
__device__ __forceinline__ void gemm_stage(
    const _Float16* __restrict__ X, const _Float16* __restrict__ W,
    _Float16 (*As)[40], _Float16 (*Bs)[40],
    int m0, int n0, int Kdim, int k0, int tid)
{
  // A tile: 128 rows x 4 chunks(8 f16) = 512 chunks, 2 per thread
  #pragma unroll
  for (int i = 0; i < 2; ++i) {
    int c = tid + i * 256;
    int row = c >> 2, cc = (c & 3) * 8;
    cp16_g2l(X + (size_t)(m0 + row) * Kdim + k0 + cc, &As[row][cc]);
  }
  // B tile: 64 rows x 4 chunks = 256 chunks, 1 per thread
  {
    int row = tid >> 2, cc = (tid & 3) * 8;
    cp16_g2l(W + (size_t)(n0 + row) * Kdim + k0 + cc, &Bs[row][cc]);
  }
}

__global__ __launch_bounds__(256) void gemm_xwt(
    const _Float16* __restrict__ X, const _Float16* __restrict__ W,
    void* __restrict__ Y, int M, int N, int Kdim, int mode)
{
  __shared__ _Float16 As[2][128][40];   // row stride 80B (16B multiple)
  __shared__ _Float16 Bs[2][64][40];

  const int tid  = threadIdx.x;
  const int lane = tid & 31;
  const int wave = tid >> 5;
  const int mw   = wave >> 1;
  const int nw   = wave & 1;
  const int m0   = blockIdx.y * 128;
  const int n0   = blockIdx.x * 64;
  const int kh   = lane >> 4;
  const int l15  = lane & 15;

  v8f zero = {};
  v8f acc[2][2];
  acc[0][0] = zero; acc[0][1] = zero; acc[1][0] = zero; acc[1][1] = zero;

  const int nk = Kdim / 32;
  gemm_stage(X, W, As[0], Bs[0], m0, n0, Kdim, 0, tid);
  int cur = 0;
  for (int it = 0; it < nk; ++it) {
    if (it + 1 < nk) {
      gemm_stage(X, W, As[cur ^ 1], Bs[cur ^ 1], m0, n0, Kdim, (it + 1) * 32, tid);
      wait_async<3>();   // 3 in flight for the next tile; current tile drained
    } else {
      wait_async<0>();
    }
    __syncthreads();

    v16h a[2], b[2];
    #pragma unroll
    for (int i = 0; i < 2; ++i) {
      int mr = mw * 32 + i * 16 + l15;
      #pragma unroll
      for (int e = 0; e < 8; ++e) {
        a[i][e]     = As[cur][mr][kh * 8 + e];
        a[i][e + 8] = As[cur][mr][16 + kh * 8 + e];
      }
    }
    #pragma unroll
    for (int j = 0; j < 2; ++j) {
      int nr = nw * 32 + j * 16 + l15;
      #pragma unroll
      for (int e = 0; e < 8; ++e) {
        b[j][e]     = Bs[cur][nr][kh * 8 + e];
        b[j][e + 8] = Bs[cur][nr][16 + kh * 8 + e];
      }
    }
    #pragma unroll
    for (int i = 0; i < 2; ++i)
      #pragma unroll
      for (int j = 0; j < 2; ++j)
        acc[i][j] = __builtin_amdgcn_wmma_f32_16x16x32_f16(
            false, a[i], false, b[j], (short)0, acc[i][j], false, false);
    __syncthreads();
    cur ^= 1;
  }

  #pragma unroll
  for (int i = 0; i < 2; ++i) {
    #pragma unroll
    for (int j = 0; j < 2; ++j) {
      #pragma unroll
      for (int r = 0; r < 8; ++r) {
        int m = m0 + mw * 32 + i * 16 + r + (kh ? 8 : 0);
        int n = n0 + nw * 32 + j * 16 + l15;
        float v = acc[i][j][r];
        if (mode == 0) {
          ((float*)Y)[(size_t)m * N + n] = v;
        } else {
          int bb = m >> 11, s = m & 2047;
          int hh = n >> 6,  d = n & 63;
          if (mode == 1)
            ((_Float16*)Y)[(((size_t)bb * N_HEAD + hh) * SEQ + s) * D_K + d] = (_Float16)v;
          else
            ((_Float16*)Y)[(((size_t)bb * N_HEAD + hh) * D_K + d) * SEQ + s] = (_Float16)v;
        }
      }
    }
  }
}

// ============================================================================
// Flash attention. Q,K: f16 [B,H,S,d_k]; V: f16 [B,H,d_k,S] (pre-transposed).
// Block 128 thr = 4 waves, 64 q-rows per block. Double-buffered async K/V
// tiles (64 keys). Online softmax; P via per-wave LDS C->A layout conversion.
// ============================================================================
__device__ __forceinline__ void attn_stage(
    const _Float16* __restrict__ Kbh, const _Float16* __restrict__ Vbh,
    _Float16 (*Kt)[72], _Float16 (*Vt)[72], int kbase, int tid)
{
  // K tile: 64 keys x 8 chunks; V^T tile: 64 d x 8 chunks; 4+4 per thread
  #pragma unroll
  for (int i = 0; i < 4; ++i) {
    int c = tid + i * 128;
    int row = c >> 3, cc = (c & 7) * 8;
    cp16_g2l(Kbh + (size_t)(kbase + row) * D_K + cc, &Kt[row][cc]);
    cp16_g2l(Vbh + (size_t)row * SEQ + kbase + cc, &Vt[row][cc]);
  }
}

__global__ __launch_bounds__(128) void flash_attn(
    const _Float16* __restrict__ Q, const _Float16* __restrict__ K,
    const _Float16* __restrict__ V, const int* __restrict__ mask,
    _Float16* __restrict__ ctx)
{
  __shared__ _Float16 Kt[2][64][72];     // [key][d], row stride 144B
  __shared__ _Float16 Vt[2][64][72];     // [d][key] (already transposed)
  __shared__ _Float16 Pl[4][16][72];     // per-wave P scratch

  const int tid  = threadIdx.x;
  const int lane = tid & 31;
  const int wave = tid >> 5;
  const int l15  = lane & 15;
  const int kh   = lane >> 4;
  const int b    = blockIdx.y >> 4;
  const int h    = blockIdx.y & 15;
  const int q0   = blockIdx.x * 64;

  const _Float16* Qbh = Q + ((size_t)b * N_HEAD + h) * SEQ * D_K;
  const _Float16* Kbh = K + ((size_t)b * N_HEAD + h) * SEQ * D_K;
  const _Float16* Vbh = V + ((size_t)b * N_HEAD + h) * D_K * SEQ;

  // Q fragments (A-matrix 16x64 = 2 k-steps), resident in VGPRs
  const int qrow = q0 + wave * 16 + l15;
  v16h qf[2];
  #pragma unroll
  for (int ks = 0; ks < 2; ++ks) {
    #pragma unroll
    for (int e = 0; e < 8; ++e) {
      qf[ks][e]     = Qbh[(size_t)qrow * D_K + ks * 32 + kh * 8 + e];
      qf[ks][e + 8] = Qbh[(size_t)qrow * D_K + ks * 32 + 16 + kh * 8 + e];
    }
  }

  v8f zero = {};
  v8f cacc[4];
  float mrow[8], ssum[8];
  #pragma unroll
  for (int t = 0; t < 4; ++t) cacc[t] = zero;
  #pragma unroll
  for (int r = 0; r < 8; ++r) { mrow[r] = -INFINITY; ssum[r] = 0.0f; }

  const int NB = SEQ / 64;
  attn_stage(Kbh, Vbh, Kt[0], Vt[0], 0, tid);
  int cur = 0;
  for (int kb = 0; kb < NB; ++kb) {
    const int kbase = kb * 64;
    if (kb + 1 < NB) {
      attn_stage(Kbh, Vbh, Kt[cur ^ 1], Vt[cur ^ 1], (kb + 1) * 64, tid);
      wait_async<8>();   // 8 in flight for next tile; current tile drained
    } else {
      wait_async<0>();
    }
    __syncthreads();

    // scores: 16q x 64 keys = 4 tiles x 2 k-steps
    v8f sacc[4];
    #pragma unroll
    for (int t = 0; t < 4; ++t) sacc[t] = zero;
    #pragma unroll
    for (int t = 0; t < 4; ++t) {
      #pragma unroll
      for (int ks = 0; ks < 2; ++ks) {
        v16h bf;
        int krow = t * 16 + l15;
        #pragma unroll
        for (int e = 0; e < 8; ++e) {
          bf[e]     = Kt[cur][krow][ks * 32 + kh * 8 + e];
          bf[e + 8] = Kt[cur][krow][ks * 32 + 16 + kh * 8 + e];
        }
        sacc[t] = __builtin_amdgcn_wmma_f32_16x16x32_f16(
            false, qf[ks], false, bf, (short)0, sacc[t], false, false);
      }
    }

    // scale + mask (reference: where(mask==0, -1e9, s/sqrt(dk)))
    #pragma unroll
    for (int t = 0; t < 4; ++t) {
      #pragma unroll
      for (int r = 0; r < 8; ++r) {
        int qq = q0 + wave * 16 + r + (kh ? 8 : 0);
        int kk = kbase + t * 16 + l15;
        float v = sacc[t][r] * 0.125f;
        if (mask[((size_t)b * SEQ + qq) * SEQ + kk] == 0) v = -1e9f;
        sacc[t][r] = v;
      }
    }

    // row max across tiles, then across 16 lanes of each half
    float mloc[8];
    #pragma unroll
    for (int r = 0; r < 8; ++r)
      mloc[r] = fmaxf(fmaxf(sacc[0][r], sacc[1][r]), fmaxf(sacc[2][r], sacc[3][r]));
    #pragma unroll
    for (int off = 1; off < 16; off <<= 1)
      #pragma unroll
      for (int r = 0; r < 8; ++r)
        mloc[r] = fmaxf(mloc[r], __shfl_xor(mloc[r], off, 32));

    float alpha[8];
    #pragma unroll
    for (int r = 0; r < 8; ++r) {
      float nm = fmaxf(mrow[r], mloc[r]);
      alpha[r] = __expf(mrow[r] - nm);
      mrow[r]  = nm;
    }
    float rs[8];
    #pragma unroll
    for (int r = 0; r < 8; ++r) rs[r] = 0.0f;
    #pragma unroll
    for (int t = 0; t < 4; ++t)
      #pragma unroll
      for (int r = 0; r < 8; ++r) {
        float p = __expf(sacc[t][r] - mrow[r]);
        sacc[t][r] = p;
        rs[r] += p;
      }
    #pragma unroll
    for (int off = 1; off < 16; off <<= 1)
      #pragma unroll
      for (int r = 0; r < 8; ++r) rs[r] += __shfl_xor(rs[r], off, 32);
    #pragma unroll
    for (int r = 0; r < 8; ++r) ssum[r] = ssum[r] * alpha[r] + rs[r];
    #pragma unroll
    for (int t = 0; t < 4; ++t)
      #pragma unroll
      for (int r = 0; r < 8; ++r) cacc[t][r] *= alpha[r];

    // P: C-layout -> LDS -> A-layout f16 fragments
    #pragma unroll
    for (int t = 0; t < 4; ++t)
      #pragma unroll
      for (int r = 0; r < 8; ++r)
        Pl[wave][r + (kh ? 8 : 0)][t * 16 + l15] = (_Float16)sacc[t][r];
    __syncthreads();

    // ctx += P @ V
    #pragma unroll
    for (int ks = 0; ks < 2; ++ks) {
      v16h pf;
      #pragma unroll
      for (int e = 0; e < 8; ++e) {
        pf[e]     = Pl[wave][l15][ks * 32 + kh * 8 + e];
        pf[e + 8] = Pl[wave][l15][ks * 32 + 16 + kh * 8 + e];
      }
      #pragma unroll
      for (int t = 0; t < 4; ++t) {
        v16h vf;
        int drow = t * 16 + l15;
        #pragma unroll
        for (int e = 0; e < 8; ++e) {
          vf[e]     = Vt[cur][drow][ks * 32 + kh * 8 + e];
          vf[e + 8] = Vt[cur][drow][ks * 32 + 16 + kh * 8 + e];
        }
        cacc[t] = __builtin_amdgcn_wmma_f32_16x16x32_f16(
            false, pf, false, vf, (short)0, cacc[t], false, false);
      }
    }
    __syncthreads();
    cur ^= 1;
  }

  // normalize, write ctx f16 in [B,S,D] layout for output GEMM
  #pragma unroll
  for (int r = 0; r < 8; ++r) {
    float inv = (ssum[r] > 0.0f) ? (1.0f / ssum[r]) : 0.0f;
    int qq = q0 + wave * 16 + r + (kh ? 8 : 0);
    #pragma unroll
    for (int t = 0; t < 4; ++t) {
      int d = h * D_K + t * 16 + l15;
      ctx[((size_t)b * SEQ + qq) * D_MODEL + d] = (_Float16)(cacc[t][r] * inv);
    }
  }
}

// ============================================================================
extern "C" void kernel_launch(void* const* d_in, const int* in_sizes, int n_in,
                              void* d_out, int out_size, void* d_ws, size_t ws_size,
                              hipStream_t stream) {
  const float* q    = (const float*)d_in[0];
  const float* k    = (const float*)d_in[1];
  const float* v    = (const float*)d_in[2];
  const float* wq   = (const float*)d_in[3];
  const float* wk   = (const float*)d_in[4];
  const float* wv   = (const float*)d_in[5];
  const float* wo   = (const float*)d_in[6];
  const int*   mask = (const int*)d_in[7];
  float*       out  = (float*)d_out;

  const size_t NX = (size_t)BATCH * SEQ * D_MODEL;   // 4,194,304
  const size_t NW = (size_t)D_MODEL * D_MODEL;       // 1,048,576

  _Float16* p    = (_Float16*)d_ws;
  _Float16* xq16 = p;            p += NX;
  _Float16* xk16 = p;            p += NX;
  _Float16* xv16 = p;            p += NX;
  _Float16* wq16 = p;            p += NW;
  _Float16* wk16 = p;            p += NW;
  _Float16* wv16 = p;            p += NW;
  _Float16* wo16 = p;            p += NW;
  _Float16* Qh   = p;            p += NX;   // [B,H,S,d_k]
  _Float16* Kh   = p;            p += NX;   // [B,H,S,d_k]
  _Float16* Vht  = p;            p += NX;   // [B,H,d_k,S]
  _Float16* Ctx  = p;            p += NX;   // [B,S,D] f16

  // fp32 -> f16 conversions
  cvt_f32_f16<<<(int)(NX / 4 / 256), 256, 0, stream>>>(q,  xq16, (int)(NX / 4));
  cvt_f32_f16<<<(int)(NX / 4 / 256), 256, 0, stream>>>(k,  xk16, (int)(NX / 4));
  cvt_f32_f16<<<(int)(NX / 4 / 256), 256, 0, stream>>>(v,  xv16, (int)(NX / 4));
  cvt_f32_f16<<<(int)(NW / 4 / 256), 256, 0, stream>>>(wq, wq16, (int)(NW / 4));
  cvt_f32_f16<<<(int)(NW / 4 / 256), 256, 0, stream>>>(wk, wk16, (int)(NW / 4));
  cvt_f32_f16<<<(int)(NW / 4 / 256), 256, 0, stream>>>(wv, wv16, (int)(NW / 4));
  cvt_f32_f16<<<(int)(NW / 4 / 256), 256, 0, stream>>>(wo, wo16, (int)(NW / 4));

  const int M = BATCH * SEQ;   // 4096
  dim3 gg(D_MODEL / 64, M / 128);      // (16, 32)
  gemm_xwt<<<gg, 256, 0, stream>>>(xq16, wq16, Qh,  M, D_MODEL, D_MODEL, 1);
  gemm_xwt<<<gg, 256, 0, stream>>>(xk16, wk16, Kh,  M, D_MODEL, D_MODEL, 1);
  gemm_xwt<<<gg, 256, 0, stream>>>(xv16, wv16, Vht, M, D_MODEL, D_MODEL, 2);

  dim3 ga(SEQ / 64, BATCH * N_HEAD);   // (32, 32)
  flash_attn<<<ga, 128, 0, stream>>>(Qh, Kh, Vht, mask, Ctx);

  gemm_xwt<<<gg, 256, 0, stream>>>(Ctx, wo16, out, M, D_MODEL, D_MODEL, 0);
}
